// PredictionLatticeCore_47639777247426
// MI455X (gfx1250) — matrix operand order repaired and checked by reading, not compiled, not run on hardware
//
#include <hip/hip_runtime.h>
#include <hip/hip_fp16.h>

typedef __attribute__((ext_vector_type(16))) _Float16 v16h;
typedef __attribute__((ext_vector_type(4)))  _Float16 v4h;
typedef __attribute__((ext_vector_type(8)))  float    v8f;

#define D_MODEL 2048
#define SEQ     8192
#define NB      2
#define NSPINE  10
#define NROWS   20   /* NB * NSPINE */
#define MPAD    32
#define LN_EPS  1e-5f

/* ---------------- workspace layout (bytes) ----------------
   hA   : MPAD   * D_MODEL * 2  = 131072      (f16 A-matrix, padded M)
   hF32 : MPAD   * D_MODEL * 4  = 262144      (f32 h rows for LN path)
   Wf16 : D_MODEL* D_MODEL * 2  = 8388608     (folded f16 weights W'[d,j]=W[d,j]+W[d,j+2048])
   gpre : MPAD   * D_MODEL * 4  = 262144      (raw gate accumulators)
   total ~ 8.6 MB                                                  */
#define OFF_HA   0
#define OFF_HF   131072
#define OFF_WF   (131072 + 262144)
#define OFF_GP   (131072 + 262144 + 8388608)

/* ------------- 1) pass-through copy: out = x (bandwidth bound) ------------- */
__global__ void plc_copy(const float4* __restrict__ src, float4* __restrict__ dst, long n4) {
    long i      = (long)blockIdx.x * blockDim.x + threadIdx.x;
    long stride = (long)gridDim.x * blockDim.x;
    for (; i < n4; i += stride) dst[i] = src[i];
}

/* ------------- 2) gather spine rows, h = 0.5*x[row]; build f16 A ------------- */
__global__ void plc_prep(const float* __restrict__ x, const int* __restrict__ spine,
                         _Float16* __restrict__ hA, float* __restrict__ hF) {
    int m = blockIdx.x; /* 0..31 */
    if (m < NROWS) {
        int b = m / NSPINE, k = m % NSPINE;
        int row = spine[k];
        const float* src = x + (long)(b * SEQ + row) * D_MODEL;
        for (int d = threadIdx.x; d < D_MODEL; d += blockDim.x) {
            float h = 0.5f * src[d];
            hF[m * D_MODEL + d] = h;
            hA[m * D_MODEL + d] = (_Float16)h;
        }
    } else { /* zero pad rows 20..31 */
        for (int d = threadIdx.x; d < D_MODEL; d += blockDim.x) {
            hF[m * D_MODEL + d] = 0.0f;
            hA[m * D_MODEL + d] = (_Float16)0.0f;
        }
    }
}

/* ------------- 3) fold concat([h,h]) into weights: W'[d,j] = W[d,j]+W[d,j+2048] ------------- */
__global__ void plc_fold(const float* __restrict__ W, _Float16* __restrict__ Wf) {
    int i      = blockIdx.x * blockDim.x + threadIdx.x;
    int total  = D_MODEL * (D_MODEL / 4);
    int stride = gridDim.x * blockDim.x;
    for (; i < total; i += stride) {
        int d  = i >> 9;             /* D_MODEL/4 = 512 quads per row */
        int jq = (i & 511) << 2;
        const float4 w0 = *reinterpret_cast<const float4*>(W + (long)d * (2 * D_MODEL) + jq);
        const float4 w1 = *reinterpret_cast<const float4*>(W + (long)d * (2 * D_MODEL) + D_MODEL + jq);
        v4h o;
        o[0] = (_Float16)(w0.x + w1.x);
        o[1] = (_Float16)(w0.y + w1.y);
        o[2] = (_Float16)(w0.z + w1.z);
        o[3] = (_Float16)(w0.w + w1.w);
        *reinterpret_cast<v4h*>(Wf + (long)d * D_MODEL + jq) = o;
    }
}

/* ------------- 4) WMMA GEMM: gpre[32,2048] = A[32,2048] * W'^T ------------- */
union AB32 { v16h h; float4 f[2]; };

__global__ void plc_gate_gemm(const _Float16* __restrict__ hA,
                              const _Float16* __restrict__ Wf,
                              float* __restrict__ gpre) {
    const int lane   = threadIdx.x & 31;
    const int wave   = blockIdx.x * (blockDim.x >> 5) + (threadIdx.x >> 5); /* 0..255 */
    const int m_tile = wave >> 7;   /* 0..1   */
    const int n_tile = wave & 127;  /* 0..127 */
    const int hi     = lane >> 4;   /* half-wave select */
    const int l16    = lane & 15;

    /* A fragment (16-bit 16x32, ISA 7.12.2): lane<16 -> M=lane, K 0-7 then 16-23;
       lane>=16 -> same M rows, K 8-15 then 24-31.  Row-major hA[m][k].            */
    const _Float16* aBase = hA + (long)(m_tile * 16 + l16) * D_MODEL + hi * 8;
    /* B fragment (32x16): lanes 0-15 hold K 0-7 then 16-23 (N=lane), lanes 16-31 hold K 8-15 then 24-31.
       W' is [N=d][K=j] row-major, so two runs of 8 consecutive halves per lane.   */
    const _Float16* bBase = Wf + (long)(n_tile * 16 + l16) * D_MODEL + hi * 8;

    v8f c = {};
    for (int kt = 0; kt < D_MODEL / 32; ++kt) {
        AB32 ua, ub;
        const float4* pa = reinterpret_cast<const float4*>(aBase + kt * 32);
        ua.f[0] = pa[0];            /* K +0..7   */
        ua.f[1] = pa[2];            /* K +16..23 */
        const float4* pb = reinterpret_cast<const float4*>(bBase + kt * 32);
        ub.f[0] = pb[0];
        ub.f[1] = pb[2];
        if (kt + 1 < D_MODEL / 32)
            __builtin_prefetch(bBase + (kt + 1) * 32, 0, 1); /* global_prefetch_b8 */
        c = __builtin_amdgcn_wmma_f32_16x16x32_f16(/*neg_a=*/false, ua.h,
                                                   /*neg_b=*/false, ub.h,
                                                   /*c_mod=*/(short)0, c,
                                                   /*reuse_a=*/false, /*reuse_b=*/false);
    }

    /* C/D layout: lane<16 -> N=lane, VGPR r = M r; lane>=16 -> M 8+r */
    const int n0    = n_tile * 16 + l16;
    const int mBase = m_tile * 16 + hi * 8;
#pragma unroll
    for (int r = 0; r < 8; ++r)
        gpre[(long)(mBase + r) * D_MODEL + n0] = c[r];
}

/* ------------- 5) gate combine + LayerNorm + scatter ------------- */
__global__ void plc_finalize(const float* __restrict__ hF,
                             const float* __restrict__ gpre,
                             const float* __restrict__ gate_b,
                             const float* __restrict__ ln_g,
                             const float* __restrict__ ln_b,
                             const int*   __restrict__ spine,
                             float* __restrict__ out) {
    __shared__ float s_sum[256];
    __shared__ float s_sq[256];
    const int m   = blockIdx.x;  /* 0..19 */
    const int b   = m / NSPINE, k = m % NSPINE;
    const int tid = threadIdx.x;

    float xo[D_MODEL / 256];
    float sum = 0.0f, sq = 0.0f;
#pragma unroll
    for (int i = 0; i < D_MODEL / 256; ++i) {
        int   d = tid + i * 256;
        float h = hF[m * D_MODEL + d];
        float z = gpre[m * D_MODEL + d] + gate_b[d];
        float g = 1.0f / (1.0f + __expf(-z));
        float v = g * h + (1.0f - g) * h;   /* == h, kept for fidelity */
        xo[i] = v;
        sum += v;
        sq  += v * v;
    }
    s_sum[tid] = sum; s_sq[tid] = sq;
    __syncthreads();
    for (int off = 128; off > 0; off >>= 1) {
        if (tid < off) { s_sum[tid] += s_sum[tid + off]; s_sq[tid] += s_sq[tid + off]; }
        __syncthreads();
    }
    const float mean = s_sum[0] * (1.0f / D_MODEL);
    const float var  = s_sq[0] * (1.0f / D_MODEL) - mean * mean; /* population var (jnp.var) */
    const float inv  = rsqrtf(var + LN_EPS);

    const int row = spine[k];
    float* dst = out + (long)(b * SEQ + row) * D_MODEL;
#pragma unroll
    for (int i = 0; i < D_MODEL / 256; ++i) {
        int d = tid + i * 256;
        dst[d] = (xo[i] - mean) * inv * ln_g[d] + ln_b[d];
    }
}

extern "C" void kernel_launch(void* const* d_in, const int* in_sizes, int n_in,
                              void* d_out, int out_size, void* d_ws, size_t ws_size,
                              hipStream_t stream) {
    const float* x      = (const float*)d_in[0];
    const float* gate_w = (const float*)d_in[1];
    const float* gate_b = (const float*)d_in[2];
    const float* ln_g   = (const float*)d_in[3];
    const float* ln_b   = (const float*)d_in[4];
    const int*   spine  = (const int*)d_in[5];
    float* out = (float*)d_out;

    char* ws = (char*)d_ws;
    _Float16* hA = (_Float16*)(ws + OFF_HA);
    float*    hF = (float*)   (ws + OFF_HF);
    _Float16* Wf = (_Float16*)(ws + OFF_WF);
    float*    gp = (float*)   (ws + OFF_GP);

    const long n4 = (long)NB * SEQ * D_MODEL / 4;   /* float4 count */
    plc_copy     <<<2048, 256, 0, stream>>>((const float4*)x, (float4*)out, n4);
    plc_prep     <<<MPAD, 256, 0, stream>>>(x, spine, hA, hF);
    plc_fold     <<<2048, 256, 0, stream>>>(gate_w, Wf);
    plc_gate_gemm<<<64, 128, 0, stream>>>(hA, Wf, gp);
    plc_finalize <<<NROWS, 256, 0, stream>>>(hF, gp, gate_b, ln_g, ln_b, spine, out);
}
